// HybridQLSTMQuantum_65481071396314
// MI455X (gfx1250) — matrix operand order repaired and checked by reading, not compiled, not run on hardware
//
#include <hip/hip_runtime.h>
#include <hip/hip_bf16.h>

typedef __attribute__((ext_vector_type(16))) _Float16 v16h;
typedef __attribute__((ext_vector_type(8)))  float    v8f;

#define S_LEN 256
#define BATCH 256
#define DIM   256
#define HID   8
#define FAN   264            // DIM + HID
#define NGATE 32             // 4 gates * 8
#define MROWS (S_LEN * BATCH)        // 65536
#define MTILES (MROWS / 16)          // 4096

// ---------------- pack kernel: W -> f16 B-fragment layout, bias+theta, Wh ----
__global__ __launch_bounds__(256) void qlstm_pack(
    const float* __restrict__ Wf, const float* __restrict__ bf,
    const float* __restrict__ Wi, const float* __restrict__ bi,
    const float* __restrict__ Wu, const float* __restrict__ bu,
    const float* __restrict__ Wo, const float* __restrict__ bo,
    const float* __restrict__ tf, const float* __restrict__ ti,
    const float* __restrict__ tu, const float* __restrict__ to,
    _Float16* __restrict__ Wpack, float* __restrict__ bias32,
    float* __restrict__ Whp)
{
    int t = blockIdx.x * blockDim.x + threadIdx.x;
    const float* Ws[4] = {Wf, Wi, Wu, Wo};
    if (t < 2 * 8 * 32 * 16) {                 // 8192 entries
        int e     = t & 15;
        int lane  = (t >> 4) & 31;
        int kk    = (t >> 9) & 7;
        int ntile = t >> 12;
        int n = ntile * 16 + (lane & 15);      // output column 0..31
        int k = kk * 32 + ((e >> 3) * 16) + ((lane >> 4) * 8) + (e & 7);  // B layout: lanes16-31 => K+8
        int g = n >> 3, j = n & 7;
        Wpack[t] = (_Float16)Ws[g][j * FAN + k];
    }
    if (t < NGATE) {
        const float* bs[4] = {bf, bi, bu, bo};
        const float* ts[4] = {tf, ti, tu, to};
        int g = t >> 3, j = t & 7;
        bias32[t] = bs[g][j] + ts[g][j];
    }
    if (t < NGATE * HID) {                     // Whp[n][jj] = Wg[j][256+jj]
        int n = t >> 3, jj = t & 7;
        int g = n >> 3, j = n & 7;
        Whp[t] = Ws[g][j * FAN + DIM + jj];
    }
}

// ---------------- big GEMM: XW[m][n] = sum_d X[m][d]*W[n][d] + bias[n] -------
// one wave per 16-row M tile; two 16x16 accumulators cover N=32; K=256 in 8 WMMAs
__global__ __launch_bounds__(256) void qlstm_gemm(
    const float* __restrict__ X,        // (65536, 256)
    const _Float16* __restrict__ Wpack, // (2,8,32,16) B-fragments
    const float* __restrict__ bias32,
    float* __restrict__ XW)             // (65536, 32)
{
    const int lane  = threadIdx.x & 31;
    const int wave  = threadIdx.x >> 5;
    const int tileM = blockIdx.x * 8 + wave;   // 0..4095
    const int mrow  = (lane & 15);             // A-frag row within tile
    const int kbase = (lane >> 4) * 8;         // A layout: lanes16-31 => K+8
    const float* arow = X + (size_t)(tileM * 16 + mrow) * DIM;

    v8f acc0 = {}, acc1 = {};
#pragma unroll
    for (int kk = 0; kk < 8; ++kk) {
        // ---- A fragment (16x32 f16): 8 float2 loads, convert to f16 ----
        v16h a;
        const float* ap = arow + kk * 32 + kbase;
#pragma unroll
        for (int p = 0; p < 4; ++p) {
            float2 lo = *(const float2*)(ap + 2 * p);        // K = kbase+2p,+1
            float2 hi = *(const float2*)(ap + 16 + 2 * p);   // K = 16+kbase+2p,+1
            a[2 * p]     = (_Float16)lo.x;  a[2 * p + 1]     = (_Float16)lo.y;
            a[8 + 2 * p] = (_Float16)hi.x;  a[8 + 2 * p + 1] = (_Float16)hi.y;
        }
        // ---- B fragments: contiguous 32B per lane (packed in frag layout) ----
        v16h b0 = *(const v16h*)(Wpack + (size_t)(((0 * 8 + kk) * 32 + lane)) * 16);
        v16h b1 = *(const v16h*)(Wpack + (size_t)(((1 * 8 + kk) * 32 + lane)) * 16);
        acc0 = __builtin_amdgcn_wmma_f32_16x16x32_f16(false, a, false, b0,
                                                      (short)0, acc0, false, false);
        acc1 = __builtin_amdgcn_wmma_f32_16x16x32_f16(false, a, false, b1,
                                                      (short)0, acc1, false, false);
    }
    // ---- bias + store. C/D layout: VGPR v -> row v + 8*(lane>=16), col lane&15
    const int n0    = lane & 15;
    const float bs0 = bias32[n0];
    const float bs1 = bias32[16 + n0];
    const int mb    = tileM * 16 + (lane >> 4) * 8;
#pragma unroll
    for (int v = 0; v < 8; ++v) {
        XW[(size_t)(mb + v) * NGATE + n0]      = acc0[v] + bs0;
        XW[(size_t)(mb + v) * NGATE + 16 + n0] = acc1[v] + bs1;
    }
}

// ---------------- recurrent scan: one thread per batch element ---------------
__device__ __forceinline__ float sigm_f(float x) {
    return 1.0f / (1.0f + __expf(-x));
}
__device__ __forceinline__ float tanh_f(float x) {
    float t = __expf(2.0f * x);
    return (t - 1.0f) / (t + 1.0f);
}

__global__ __launch_bounds__(32) void qlstm_scan(
    const float* __restrict__ XW,   // (S*B, 32) pre-activations (x part + bias)
    const float* __restrict__ Whp,  // (32, 8) recurrent weights
    const float* __restrict__ hx, const float* __restrict__ cx,
    float* __restrict__ out)        // outs (S,B,8) ++ hT (B,8) ++ cT (B,8)
{
    __shared__ float Wh[NGATE * HID];
    for (int i = threadIdx.x; i < NGATE * HID; i += 32) Wh[i] = Whp[i];
    __syncthreads();

    const int b = blockIdx.x * 32 + threadIdx.x;   // 0..255
    float h[HID], c[HID];
#pragma unroll
    for (int w = 0; w < HID; ++w) { h[w] = hx[b * HID + w]; c[w] = cx[b * HID + w]; }

    const float* xwB = XW + (size_t)b * NGATE;
    for (int s = 0; s < S_LEN; ++s) {
        const float4* xw4 = (const float4*)(xwB + (size_t)s * BATCH * NGATE);
        float pre[NGATE];
#pragma unroll
        for (int q = 0; q < 8; ++q) {
            float4 v = xw4[q];
            pre[4 * q + 0] = v.x; pre[4 * q + 1] = v.y;
            pre[4 * q + 2] = v.z; pre[4 * q + 3] = v.w;
        }
#pragma unroll
        for (int n = 0; n < NGATE; ++n) {
            const float4* w4 = (const float4*)(&Wh[n * HID]);
            float4 w0 = w4[0], w1 = w4[1];
            float acc = pre[n];
            acc = fmaf(h[0], w0.x, acc); acc = fmaf(h[1], w0.y, acc);
            acc = fmaf(h[2], w0.z, acc); acc = fmaf(h[3], w0.w, acc);
            acc = fmaf(h[4], w1.x, acc); acc = fmaf(h[5], w1.y, acc);
            acc = fmaf(h[6], w1.z, acc); acc = fmaf(h[7], w1.w, acc);
            pre[n] = acc;
        }
        // qgate closed form: <Z_w> = prod_{j<=w} cos(pre[j]) per gate
        float ev[NGATE];
#pragma unroll
        for (int g = 0; g < 4; ++g) {
            float p = 1.0f;
#pragma unroll
            for (int w = 0; w < HID; ++w) { p *= __cosf(pre[g * HID + w]); ev[g * HID + w] = p; }
        }
        float hv[HID];
#pragma unroll
        for (int w = 0; w < HID; ++w) {
            float fg = sigm_f(ev[w]);
            float ig = sigm_f(ev[HID + w]);
            float gg = tanh_f(ev[2 * HID + w]);
            float og = sigm_f(ev[3 * HID + w]);
            c[w] = fg * c[w] + ig * gg;
            hv[w] = og * tanh_f(c[w]);
            h[w] = hv[w];
        }
        float4* o4 = (float4*)(out + (size_t)s * BATCH * HID + (size_t)b * HID);
        o4[0] = make_float4(hv[0], hv[1], hv[2], hv[3]);
        o4[1] = make_float4(hv[4], hv[5], hv[6], hv[7]);
    }
    float* hT = out + (size_t)S_LEN * BATCH * HID;
    float* cT = hT + (size_t)BATCH * HID;
#pragma unroll
    for (int w = 0; w < HID; ++w) { hT[b * HID + w] = h[w]; cT[b * HID + w] = c[w]; }
}

// ---------------- launch ----------------
extern "C" void kernel_launch(void* const* d_in, const int* in_sizes, int n_in,
                              void* d_out, int out_size, void* d_ws, size_t ws_size,
                              hipStream_t stream) {
    const float* inputs = (const float*)d_in[0];
    const float* hx = (const float*)d_in[1];
    const float* cx = (const float*)d_in[2];
    const float* Wf = (const float*)d_in[3];  const float* bf = (const float*)d_in[4];
    const float* Wi = (const float*)d_in[5];  const float* bi = (const float*)d_in[6];
    const float* Wu = (const float*)d_in[7];  const float* bu = (const float*)d_in[8];
    const float* Wo = (const float*)d_in[9];  const float* bo = (const float*)d_in[10];
    const float* tf = (const float*)d_in[11]; const float* ti = (const float*)d_in[12];
    const float* tu = (const float*)d_in[13]; const float* to = (const float*)d_in[14];

    char* ws = (char*)d_ws;
    float*    XW     = (float*)ws;                         // 65536*32*4 = 8 MB
    _Float16* Wpack  = (_Float16*)(ws + 8388608);          // 8192 * 2B = 16 KB
    float*    bias32 = (float*)(ws + 8388608 + 16384);     // 128 B
    float*    Whp    = (float*)(ws + 8388608 + 16384 + 128);

    qlstm_pack<<<32, 256, 0, stream>>>(Wf, bf, Wi, bi, Wu, bu, Wo, bo,
                                       tf, ti, tu, to, Wpack, bias32, Whp);
    qlstm_gemm<<<MTILES / 8, 256, 0, stream>>>(inputs, Wpack, bias32, XW);
    qlstm_scan<<<BATCH / 32, 32, 0, stream>>>(XW, Whp, hx, cx, (float*)d_out);
}
